// DGN9_70428873720413
// MI455X (gfx1250) — matrix operand order repaired and checked by reading, not compiled, not run on hardware
//
#include <hip/hip_runtime.h>
#include <hip/hip_bf16.h>
#include <math.h>

typedef __attribute__((ext_vector_type(16))) __bf16 v16bf;
typedef __attribute__((ext_vector_type(8)))  __bf16 v8bf;
typedef __attribute__((ext_vector_type(4)))  __bf16 v4bf;
typedef __attribute__((ext_vector_type(8)))  float  v8f;
typedef __attribute__((ext_vector_type(4)))  int    v4i;

#define BX 4
#define TX 4096
#define DX 1024
#define KX 8
#define TILE 128
#define KC 128
#define NEGV -1.0e30f

// CDNA5 async global->LDS path (no VGPR round-trip, tracked by ASYNCcnt).
#if defined(__gfx1250__) && \
    __has_builtin(__builtin_amdgcn_global_load_async_to_lds_b128) && \
    __has_builtin(__builtin_amdgcn_s_wait_asynccnt)
#define ASYNC_LDS 1
#define GPTR(p) ((__attribute__((address_space(1))) v4i*)(p))
#define LPTR(p) ((__attribute__((address_space(3))) v4i*)(p))
#else
#define ASYNC_LDS 0
#endif

// ---------------- fp32 -> bf16 convert (feeds WMMA operands) ----------------
__global__ void cvt_bf16_kernel(const float4* __restrict__ x,
                                v4bf* __restrict__ xb, int n4) {
  int i = blockIdx.x * blockDim.x + threadIdx.x;
  int s = gridDim.x * blockDim.x;
  for (; i < n4; i += s) {
    float4 f = x[i];
    v4bf o;
    o[0] = (__bf16)f.x; o[1] = (__bf16)f.y;
    o[2] = (__bf16)f.z; o[3] = (__bf16)f.w;
    xb[i] = o;
  }
}

// ---------------- fused causal top-K graph aggregation ----------------
// One block = one (batch, 128-row tile). 8 waves; wave w owns rows
// [w*16, w*16+16) of the score tile, 8 x (16x16) accumulators across N.
// Double-buffered async-to-LDS staging overlaps the next 128x128 bf16
// A/B chunks with the current 32-WMMA compute block.
extern "C" __global__ void __launch_bounds__(256)
topk_graph_kernel(const float* __restrict__ x,
                  const __bf16* __restrict__ xb,
                  const float* __restrict__ w_gate,
                  const float* __restrict__ b_gate,
                  const float* __restrict__ gain,
                  const float* __restrict__ bias,
                  const float* __restrict__ log_mix,
                  const float* __restrict__ log_scale,
                  float* __restrict__ out)
{
  extern __shared__ char smem[];
  // layout: As[2] @ 0, 32768 ; Bs[2] @ 65536, 98304 ; Sc @ 131072 ; topk meta
  float*  Sc   = (float*)(smem + 131072);                    // 128 x 129 f32
  int*    tIdx = (int*)(smem + 131072 + 128 * 129 * 4);      // 128 x 8
  int*    cntA = (int*)(smem + 131072 + 128 * 129 * 4 + 128 * 8 * 4);

  const int tid  = threadIdx.x;
  const int wave = tid >> 5;
  const int lane = tid & 31;
  const int hl   = lane >> 4;   // lane half (WMMA row/K split)
  const int l15  = lane & 15;

  const int b  = blockIdx.x >> 5;            // T/TILE == 32 tiles
  const int i0 = (blockIdx.x & 31) * TILE;

  const __bf16* xbB = xb + (size_t)b * TX * DX;
  const float*  xB  = x  + (size_t)b * TX * DX;

  // per-row top-K state: thread r (< 128) owns row i0+r, sorted descending
  float tv[KX];
  int   tix[KX];
#pragma unroll
  for (int k = 0; k < KX; ++k) { tv[k] = NEGV; tix[k] = -1; }

  int j0 = 0;

  // stage one 128x128 bf16 chunk of A (rows i0..) and B (rows j0..) into buf
  auto stage = [&](int buf, int kc) {
    __bf16* Ad = (__bf16*)(smem + (size_t)buf * 32768);
    __bf16* Bd = (__bf16*)(smem + 65536 + (size_t)buf * 32768);
#pragma unroll
    for (int it = 0; it < 8; ++it) {
      int ch  = it * 256 + tid;        // 0..2047 sixteen-byte chunks
      int row = ch >> 4;
      int cc  = (ch & 15) << 3;        // bf16 column (x8 per chunk)
      const __bf16* ga = xbB + (size_t)(i0 + row) * DX + kc + cc;
      const __bf16* gb = xbB + (size_t)(j0 + row) * DX + kc + cc;
#if ASYNC_LDS
      __builtin_amdgcn_global_load_async_to_lds_b128(
          GPTR(ga), LPTR(Ad + row * KC + cc), 0, 0);
      __builtin_amdgcn_global_load_async_to_lds_b128(
          GPTR(gb), LPTR(Bd + row * KC + cc), 0, 0);
#else
      *(uint4*)(Ad + row * KC + cc) = *(const uint4*)ga;
      *(uint4*)(Bd + row * KC + cc) = *(const uint4*)gb;
#endif
    }
  };
  auto wait_stage = [&]() {
#if ASYNC_LDS
    __builtin_amdgcn_s_wait_asynccnt(0);
#endif
    __syncthreads();
  };

  for (j0 = 0; j0 <= i0; j0 += TILE) {
    v8f acc[8];
#pragma unroll
    for (int n = 0; n < 8; ++n) acc[n] = (v8f){0.f,0.f,0.f,0.f,0.f,0.f,0.f,0.f};

    int cur = 0;
    stage(0, 0);
    for (int kci = 0; kci < DX / KC; ++kci) {
      wait_stage();                               // buf[cur] ready, all waves synced
      if (kci + 1 < DX / KC) stage(cur ^ 1, (kci + 1) * KC);  // prefetch next

      const __bf16* Ac = (const __bf16*)(smem + (size_t)cur * 32768);
      const __bf16* Bc = (const __bf16*)(smem + 65536 + (size_t)cur * 32768);
#pragma unroll
      for (int kk = 0; kk < KC / 32; ++kk) {
        // A fragment (16x32 bf16): lanes 0-15 rows M=l, K 0..7 | 16..23;
        // lanes 16-31 rows M=l-16, K 8..15 | 24..31  (ISA 7.12.2)
        const __bf16* pa = Ac + ((wave << 4) + l15) * KC + (kk << 5) + (hl << 3);
        v8bf a0 = *(const v8bf*)pa;
        v8bf a1 = *(const v8bf*)(pa + 16);
        v16bf av = __builtin_shufflevector(a0, a1,
            0,1,2,3,4,5,6,7,8,9,10,11,12,13,14,15);
#pragma unroll
        for (int n = 0; n < 8; ++n) {
          // B fragment (32x16 bf16): col N=l15, lanes 0-15 K 0..7 | 16..23, lanes 16-31 K 8..15 | 24..31
          const __bf16* pb = Bc + ((n << 4) + l15) * KC + (kk << 5) + (hl << 3);
          v8bf b0 = *(const v8bf*)pb;
          v8bf b1 = *(const v8bf*)(pb + 16);
          v16bf bv = __builtin_shufflevector(b0, b1,
              0,1,2,3,4,5,6,7,8,9,10,11,12,13,14,15);
          acc[n] = __builtin_amdgcn_wmma_f32_16x16x32_bf16(
              false, av, false, bv, (short)0, acc[n], false, false);
        }
      }
      cur ^= 1;
    }
    __syncthreads();
    // scatter score tile to LDS (C layout: lane<16 -> M=r, lane>=16 -> M=r+8)
#pragma unroll
    for (int n = 0; n < 8; ++n) {
#pragma unroll
      for (int r = 0; r < 8; ++r) {
        int row = (wave << 4) + (hl << 3) + r;
        int col = (n << 4) + l15;
        Sc[row * 129 + col] = acc[n][r];
      }
    }
    __syncthreads();
    // streaming top-K merge (causal mask enforced here)
    if (tid < TILE) {
      int r = tid;
      int jmax = i0 + r - j0;                       // allow c <= jmax
      int cend = (jmax < TILE - 1) ? (jmax + 1) : TILE;
      for (int c = 0; c < cend; ++c) {
        float v = Sc[r * 129 + c];
        if (v > tv[KX - 1]) {
          tv[KX - 1] = v; tix[KX - 1] = j0 + c;
#pragma unroll
          for (int k = KX - 1; k > 0; --k) {
            if (tv[k] > tv[k - 1]) {
              float tf = tv[k]; tv[k] = tv[k - 1]; tv[k - 1] = tf;
              int   tj = tix[k]; tix[k] = tix[k - 1]; tix[k - 1] = tj;
            }
          }
        }
      }
    }
  }

  // publish indices + counts
  if (tid < TILE) {
    int c = 0;
#pragma unroll
    for (int k = 0; k < KX; ++k) { tIdx[tid * KX + k] = tix[k]; c += (tix[k] >= 0); }
    cntA[tid] = (c < 1) ? 1 : c;
  }
  __syncthreads();

  // epilogue: gather-mean + gated exact-GELU. Wave w handles rows w*16..w*16+15.
  const float lm = log_mix[0];
  const float ls = log_scale[0];
  const float mix   = 1.f / (1.f + expf(-lm));
  const float scale = log1pf(expf(ls)) + 0.01f;
  const float bg    = b_gate[0];

  for (int rr = 0; rr < 16; ++rr) {
    int r = (wave << 4) + rr;
    int t = i0 + r;
    const float* xr = xB + (size_t)t * DX;

    // gate = sigmoid(x_t . w_gate + b)
    float gs = 0.f;
    for (int d = lane; d < DX; d += 32) gs += xr[d] * w_gate[d];
#pragma unroll
    for (int o = 16; o > 0; o >>= 1) gs += __shfl_xor(gs, o, 32);
    float gate = 1.f / (1.f + expf(-(gs + bg)));

    int idxs[KX];
#pragma unroll
    for (int k = 0; k < KX; ++k) idxs[k] = tIdx[r * KX + k];
    float inv = 1.f / (float)cntA[r];

    for (int d = lane; d < DX; d += 32) {
      float xv = xr[d];
      float m = 0.f;
#pragma unroll
      for (int k = 0; k < KX; ++k) {
        int j = idxs[k];
        if (j >= 0) m += xB[(size_t)j * DX + d];
      }
      float msg = m * inv;
      float bl  = mix * xv + (1.f - mix) * msg;
      float u   = bl * gain[d] + bias[d];
      float ge  = 0.5f * u * (1.f + erff(u * 0.70710678118f));
      out[((size_t)b * TX + t) * DX + d] = gate * ge * scale;
    }
  }
}

extern "C" void kernel_launch(void* const* d_in, const int* in_sizes, int n_in,
                              void* d_out, int out_size, void* d_ws, size_t ws_size,
                              hipStream_t stream) {
  const float* x         = (const float*)d_in[0];
  const float* w_gate    = (const float*)d_in[1];
  const float* b_gate    = (const float*)d_in[2];
  const float* gain      = (const float*)d_in[3];
  const float* bias      = (const float*)d_in[4];
  const float* log_mix   = (const float*)d_in[5];
  const float* log_scale = (const float*)d_in[6];
  // d_in[7] = K (== 8, hard-coded as KX)

  __bf16* xb = (__bf16*)d_ws;                 // 32 MB bf16 copy of x
  int n4 = (BX * TX * DX) / 4;
  cvt_bf16_kernel<<<2048, 256, 0, stream>>>((const float4*)x, (v4bf*)xb, n4);

  dim3 grid(BX * (TX / TILE));
  size_t shmem = 131072 + (size_t)128 * 129 * 4 + 128 * 8 * 4 + 128 * 4;
  topk_graph_kernel<<<grid, 256, shmem, stream>>>(
      x, xb, w_gate, b_gate, gain, bias, log_mix, log_scale, (float*)d_out);
}
